// CopyGenerator_41970420417302
// MI455X (gfx1250) — compile-verified, matrix operand order checked by reading, not run on hardware
//
#include <hip/hip_runtime.h>
#include <hip/hip_bf16.h>
#include <math.h>

#define TB    1600          // T*B rows
#define DH    1024          // hidden dim
#define VSZ   50257         // target vocab
#define CSZ   400           // extended (copy) vocab
#define SSZ   400           // source positions
#define TT    50
#define BB    32
#define OUTW  (VSZ + CSZ)   // 50657
#define PAD_IDX 1

typedef __bf16 bf16_t;
typedef __attribute__((ext_vector_type(16))) __bf16 v16bf;
typedef __attribute__((ext_vector_type(8)))  __bf16 v8bf;
typedef __attribute__((ext_vector_type(8)))  float  v8f;

// ---------------------------------------------------------------------------
// Kernel 1: p_copy = sigmoid(hidden . w_copy + b_copy); hidden -> bf16 in ws
// ---------------------------------------------------------------------------
__global__ __launch_bounds__(256) void prep_kernel(
    const float* __restrict__ hidden, const float* __restrict__ w_copy,
    const float* __restrict__ b_copy, bf16_t* __restrict__ hbf,
    float* __restrict__ pcopy) {
  __shared__ float red[256];
  const int r   = blockIdx.x;
  const int tid = threadIdx.x;
  const float* hrow = hidden + (size_t)r * DH;
  bf16_t*      orow = hbf    + (size_t)r * DH;
  float acc = 0.f;
  for (int k = tid; k < DH; k += 256) {
    float h = hrow[k];
    orow[k] = (bf16_t)h;
    acc += h * w_copy[k];
  }
  red[tid] = acc;
  __syncthreads();
  for (int s = 128; s > 0; s >>= 1) {
    if (tid < s) red[tid] += red[tid + s];
    __syncthreads();
  }
  if (tid == 0) pcopy[r] = 1.f / (1.f + expf(-(red[0] + b_copy[0])));
}

// ---------------------------------------------------------------------------
// Kernel 2: logits[:, :V] = hidden_bf16 @ W^T + b  (WMMA bf16, NT layout)
// Block = 128 threads = 4 waves; each wave owns one 16-col tile of a 64-col
// N-strip and loops over all 100 M-tiles, so the W strip stays hot in L2.
// A tile (16x1024 bf16) staged in LDS with padded stride (bank-friendly).
// ---------------------------------------------------------------------------
#define LDA 1032   // padded A row stride in bf16 elements (2064 B)

__global__ __launch_bounds__(128) void logits_kernel(
    const bf16_t* __restrict__ hbf, const float* __restrict__ W,
    const float* __restrict__ bias, float* __restrict__ out) {
  __shared__ bf16_t Alds[16 * LDA];   // 33,024 B
  const int tid  = threadIdx.x;
  const int wave = tid >> 5;
  const int lane = tid & 31;
  const int rr   = lane & 15;
  const int half = lane >> 4;

  const int ntile = blockIdx.x * 4 + wave;
  const int n     = ntile * 16 + rr;              // column this lane owns
  const bool nvalid = (n < VSZ);
  const int nc = nvalid ? n : (VSZ - 1);          // clamp for safe loads
  const float bn = bias[nc];
  const float* wrow = W + (size_t)nc * DH;

  for (int m0 = 0; m0 < TB; m0 += 16) {
    __syncthreads();
    {   // cooperative stage of A tile [16 x 1024] bf16 (as uint pairs)
      const unsigned* src = (const unsigned*)(hbf + (size_t)m0 * DH);
      unsigned*       dst = (unsigned*)Alds;
      for (int idx = tid; idx < 16 * (DH / 2); idx += 128) {
        int row = idx >> 9;        // 512 uints per source row
        int col = idx & 511;
        dst[row * (LDA / 2) + col] = src[row * (DH / 2) + col];
      }
    }
    __syncthreads();

    v8f acc = {};
#pragma unroll 4
    for (int k0 = 0; k0 < DH; k0 += 32) {
      // A fragment: lane rr = row, elems 0..7 -> K = k0+8*half+i,
      //             elems 8..15 -> K = k0+16+8*half+i  (ISA 16-bit A layout)
      v16bf a;
      {
        const bf16_t* pa = Alds + rr * LDA + k0 + 8 * half;
        uint4 u0 = *(const uint4*)(pa);
        uint4 u1 = *(const uint4*)(pa + 16);
        v8bf lo = __builtin_bit_cast(v8bf, u0);
        v8bf hi = __builtin_bit_cast(v8bf, u1);
#pragma unroll
        for (int i = 0; i < 8; ++i) { a[i] = lo[i]; a[8 + i] = hi[i]; }
      }
      // B fragment: NT GEMM -> same K-major per-lane pattern from W row `nc`
      v16bf bfr;
      {
        const float* pb = wrow + k0 + 8 * half;
        float4 f0 = *(const float4*)(pb);
        float4 f1 = *(const float4*)(pb + 4);
        float4 f2 = *(const float4*)(pb + 16);
        float4 f3 = *(const float4*)(pb + 20);
        bfr[0]  = (bf16_t)f0.x; bfr[1]  = (bf16_t)f0.y;
        bfr[2]  = (bf16_t)f0.z; bfr[3]  = (bf16_t)f0.w;
        bfr[4]  = (bf16_t)f1.x; bfr[5]  = (bf16_t)f1.y;
        bfr[6]  = (bf16_t)f1.z; bfr[7]  = (bf16_t)f1.w;
        bfr[8]  = (bf16_t)f2.x; bfr[9]  = (bf16_t)f2.y;
        bfr[10] = (bf16_t)f2.z; bfr[11] = (bf16_t)f2.w;
        bfr[12] = (bf16_t)f3.x; bfr[13] = (bf16_t)f3.y;
        bfr[14] = (bf16_t)f3.z; bfr[15] = (bf16_t)f3.w;
      }
      acc = __builtin_amdgcn_wmma_f32_16x16x32_bf16(
          false, a, false, bfr, (short)0, acc, false, false);
    }

    if (nvalid) {   // D layout: lane = col; VGPR r = row 8*half + r
#pragma unroll
      for (int r2 = 0; r2 < 8; ++r2) {
        int m = m0 + 8 * half + r2;
        out[(size_t)m * OUTW + n] = acc[r2] + bn;
      }
    }
  }
}

// ---------------------------------------------------------------------------
// Kernel 3: in-place row softmax over cols [0,V), PAD masked, * (1 - p_copy)
// One block per row; three sweeps, row stays hot in L2 between them.
// ---------------------------------------------------------------------------
__global__ __launch_bounds__(256) void softmax_kernel(
    float* __restrict__ out, const float* __restrict__ pcopy) {
  __shared__ float red[256];
  __shared__ float s_mx, s_sum;
  const int r   = blockIdx.x;
  const int tid = threadIdx.x;
  float* row = out + (size_t)r * OUTW;

  float mx = -INFINITY;
  for (int nn = tid; nn < VSZ; nn += 256) {
    float x = (nn == PAD_IDX) ? -INFINITY : row[nn];
    mx = fmaxf(mx, x);
  }
  red[tid] = mx; __syncthreads();
  for (int s = 128; s > 0; s >>= 1) {
    if (tid < s) red[tid] = fmaxf(red[tid], red[tid + s]);
    __syncthreads();
  }
  if (tid == 0) s_mx = red[0];
  __syncthreads();
  mx = s_mx;

  float sum = 0.f;
  for (int nn = tid; nn < VSZ; nn += 256)
    if (nn != PAD_IDX) sum += expf(row[nn] - mx);
  red[tid] = sum; __syncthreads();
  for (int s = 128; s > 0; s >>= 1) {
    if (tid < s) red[tid] += red[tid + s];
    __syncthreads();
  }
  if (tid == 0) s_sum = red[0];
  __syncthreads();

  const float scale = (1.f - pcopy[r]) / s_sum;
  for (int nn = tid; nn < VSZ; nn += 256)
    row[nn] = (nn == PAD_IDX) ? 0.f : expf(row[nn] - mx) * scale;
}

// ---------------------------------------------------------------------------
// Kernel 4: copy branch.  Per batch b: C[t,c] = sum_s (attn[t*B+b,s]*pc) *
// src_map[s,b,c], written to out[:, V:V+C].  One wave per 16x16 tile,
// WMMA bf16 over K=400 (12*32 + 16, zero-padded remainder).
// ---------------------------------------------------------------------------
__global__ __launch_bounds__(128) void copy_kernel(
    const float* __restrict__ attn, const float* __restrict__ smap,
    const float* __restrict__ pcopy, float* __restrict__ out) {
  const int tid  = threadIdx.x;
  const int wave = tid >> 5;
  const int lane = tid & 31;
  const int rr   = lane & 15;
  const int half = lane >> 4;

  const int tile = blockIdx.x * 4 + wave;   // 0 .. 3199
  const int b    = tile / 100;              // 4 m-tiles * 25 n-tiles
  const int rem  = tile % 100;
  const int mt   = rem / 25;
  const int nt   = rem % 25;

  const int  t    = mt * 16 + rr;
  const bool tval = (t < TT);
  const int  rowi = (tval ? t : (TT - 1)) * BB + b;
  const float gate = pcopy[rowi];
  const float* arow = attn + (size_t)rowi * SSZ;
  const int c = nt * 16 + rr;               // always < 400

  v8f acc = {};
  for (int k0 = 0; k0 < SSZ; k0 += 32) {    // 13 chunks, last zero-padded
    v16bf a, bfr;
#pragma unroll
    for (int i = 0; i < 8; ++i) {
      int s0 = k0 + 8 * half + i;
      int s1 = k0 + 16 + 8 * half + i;
      int s0c = (s0 < SSZ) ? s0 : (SSZ - 1);
      int s1c = (s1 < SSZ) ? s1 : (SSZ - 1);
      float a0 = arow[s0c] * gate;
      float a1 = arow[s1c] * gate;
      a[i]     = (bf16_t)((tval && s0 < SSZ) ? a0 : 0.f);
      a[8 + i] = (bf16_t)((tval && s1 < SSZ) ? a1 : 0.f);
      float b0 = smap[((size_t)s0c * BB + b) * CSZ + c];
      float b1 = smap[((size_t)s1c * BB + b) * CSZ + c];
      bfr[i]     = (bf16_t)((s0 < SSZ) ? b0 : 0.f);
      bfr[8 + i] = (bf16_t)((s1 < SSZ) ? b1 : 0.f);
    }
    acc = __builtin_amdgcn_wmma_f32_16x16x32_bf16(
        false, a, false, bfr, (short)0, acc, false, false);
  }
#pragma unroll
  for (int r2 = 0; r2 < 8; ++r2) {
    int t2 = mt * 16 + 8 * half + r2;
    if (t2 < TT)
      out[((size_t)t2 * BB + b) * OUTW + VSZ + c] = acc[r2];
  }
}

// ---------------------------------------------------------------------------
extern "C" void kernel_launch(void* const* d_in, const int* in_sizes, int n_in,
                              void* d_out, int out_size, void* d_ws,
                              size_t ws_size, hipStream_t stream) {
  const float* hidden = (const float*)d_in[0];
  const float* attn   = (const float*)d_in[1];
  const float* smap   = (const float*)d_in[2];
  const float* W      = (const float*)d_in[3];
  const float* bias   = (const float*)d_in[4];
  const float* w_copy = (const float*)d_in[5];
  const float* b_copy = (const float*)d_in[6];
  float* out = (float*)d_out;

  // workspace: hidden_bf16 (3.125 MiB) + p_copy (6.25 KiB)
  bf16_t* hbf   = (bf16_t*)d_ws;
  float*  pcopy = (float*)((char*)d_ws + (size_t)TB * DH * sizeof(bf16_t));

  prep_kernel<<<TB, 256, 0, stream>>>(hidden, w_copy, b_copy, hbf, pcopy);

  const int ntiles  = (VSZ + 15) / 16;     // 3142
  const int nblocks = (ntiles + 3) / 4;    // 786 (64-col strips)
  logits_kernel<<<nblocks, 128, 0, stream>>>(hbf, W, bias, out);

  softmax_kernel<<<TB, 256, 0, stream>>>(out, pcopy);

  copy_kernel<<<(BB * 4 * 25) / 4, 128, 0, stream>>>(attn, smap, pcopy, out);
}
